// RobustVQVAE_41549513621536
// MI455X (gfx1250) — compile-verified, matrix operand order checked by reading, not run on hardware
//
#include <hip/hip_runtime.h>
#include <hip/hip_bf16.h>
#include <math.h>

// ---------------- problem constants ----------------
constexpr int BN_   = 32768;   // batch
constexpr int K1    = 1420;    // flattened input features
constexpr int K1P   = 1440;    // padded to multiple of 32 (45*32), rows stay 16B aligned
constexpr int N1    = 512;
constexpr int N2    = 256;
constexpr int DD    = 128;
constexpr int KCODE = 4096;

// d_out layout (floats), concatenated in reference return order
constexpr size_t OFF_VEL = 0;                       // [B,2]
constexpr size_t OFF_ZE  = OFF_VEL + (size_t)BN_*2; // [B,128]
constexpr size_t OFF_ZQ  = OFF_ZE  + (size_t)BN_*DD;
constexpr size_t OFF_IDX = OFF_ZQ  + (size_t)BN_*DD;
constexpr size_t OFF_CL  = OFF_IDX + (size_t)BN_;
constexpr size_t OFF_PP  = OFF_CL + 1;
constexpr size_t OFF_NA  = OFF_PP + 1;

typedef __attribute__((ext_vector_type(16))) __bf16 v16bf;
typedef __attribute__((ext_vector_type(8)))  float  v8f;

union FragU { uint4 q[2]; v16bf v; };

// ---- CDNA5 async global->LDS copy (ASYNCcnt path), per-lane 16B ----
// Generic LDS pointer low 32 bits == DS-relative LDS address (aperture in high bits).
__device__ __forceinline__ void async_b128(void* lds_ptr, const void* gptr) {
  uint32_t lds_off = (uint32_t)(uintptr_t)lds_ptr;
  asm volatile("global_load_async_to_lds_b128 %0, %1, off"
               :: "v"(lds_off), "v"(gptr) : "memory");
}
__device__ __forceinline__ void wait_async0() {
  asm volatile("s_wait_asynccnt 0x0" ::: "memory");
}
// async completions are in-order per wave: waiting <=4 retires the previous
// tile's 4 copies while the next tile's 4 stay in flight (double buffering).
__device__ __forceinline__ void wait_async4() {
  asm volatile("s_wait_asynccnt 0x4" ::: "memory");
}

// Gather one 16x32 bf16 WMMA operand fragment from an LDS tile laid out
// row-major [row][k] with `stride` elements per row.
// ISA 16-bit A layout: lanes 0-15 = rows, hold K {0..7, 16..23};
// lanes 16-31 = same rows, hold K {8..15, 24..31}.  (B gathered from B^T rows.)
__device__ __forceinline__ v16bf ld_frag(const __bf16* base, int row, int kofs,
                                         int stride, int lane) {
  int r = row + (lane & 15);
  int h = (lane >> 4) * 8;
  FragU f;
  f.q[0] = *(const uint4*)(base + r * stride + kofs + h);
  f.q[1] = *(const uint4*)(base + r * stride + kofs + 16 + h);
  return f.v;
}

// ---------------- elementwise / prep kernels ----------------
__global__ void pack_x_kernel(const float* __restrict__ x, __bf16* __restrict__ xb) {
  int n = BN_ * K1P;
  for (int i = blockIdx.x * blockDim.x + threadIdx.x; i < n; i += gridDim.x * blockDim.x) {
    int r = i / K1P, c = i - r * K1P;
    xb[i] = (c < K1) ? (__bf16)x[(size_t)r * K1 + c] : (__bf16)0.0f;
  }
}

// cast f32 -> bf16 (flat)
__global__ void cast_kernel(const float* __restrict__ src, __bf16* __restrict__ dst, int n) {
  for (int i = blockIdx.x * blockDim.x + threadIdx.x; i < n; i += gridDim.x * blockDim.x)
    dst[i] = (__bf16)src[i];
}

// cast + transpose weights: src[K,N] f32 -> dst[N,Kp] bf16, zero-padding k in [K,Kp)
__global__ void cast_transpose_kernel(const float* __restrict__ src, __bf16* __restrict__ dst,
                                      int K, int N, int Kp) {
  int total = N * Kp;
  for (int i = blockIdx.x * blockDim.x + threadIdx.x; i < total; i += gridDim.x * blockDim.x) {
    int n = i / Kp, k = i - n * Kp;
    dst[i] = (k < K) ? (__bf16)src[(size_t)k * N + n] : (__bf16)0.0f;
  }
}

__global__ void zero_f32_kernel(float* __restrict__ p, int n) {
  for (int i = blockIdx.x * blockDim.x + threadIdx.x; i < n; i += gridDim.x * blockDim.x)
    p[i] = 0.0f;
}

__global__ void emb_norms_kernel(const float* __restrict__ emb, float* __restrict__ en) {
  int c = blockIdx.x * blockDim.x + threadIdx.x;
  if (c >= KCODE) return;
  const float* r = emb + (size_t)c * DD;
  float s = 0.f;
  for (int k = 0; k < DD; ++k) { float v = r[k]; s += v * v; }
  en[c] = s;
}

// wave-per-row LayerNorm + exact GELU, f32 in -> bf16 out. 8 rows / block (wave32).
__global__ void ln_gelu_kernel(const float* __restrict__ in, const float* __restrict__ g,
                               const float* __restrict__ be, __bf16* __restrict__ out, int N) {
  int row  = blockIdx.x * 8 + (threadIdx.x >> 5);
  int lane = threadIdx.x & 31;
  const float* xr = in + (size_t)row * N;
  float s = 0.f, sq = 0.f;
  for (int c = lane; c < N; c += 32) { float v = xr[c]; s += v; sq += v * v; }
  for (int m = 16; m >= 1; m >>= 1) { s += __shfl_xor(s, m, 32); sq += __shfl_xor(sq, m, 32); }
  float mean = s / N;
  float var  = sq / N - mean * mean;
  float rstd = rsqrtf(var + 1e-5f);
  __bf16* orow = out + (size_t)row * N;
  for (int c = lane; c < N; c += 32) {
    float y = (xr[c] - mean) * rstd * g[c] + be[c];
    float gl = 0.5f * y * (1.0f + erff(y * 0.70710678118654752f));
    orow[c] = (__bf16)gl;
  }
}

__global__ void gelu_cast_kernel(const float* __restrict__ in, __bf16* __restrict__ out, int n) {
  for (int i = blockIdx.x * blockDim.x + threadIdx.x; i < n; i += gridDim.x * blockDim.x) {
    float y = in[i];
    out[i] = (__bf16)(0.5f * y * (1.0f + erff(y * 0.70710678118654752f)));
  }
}

// ---------------- WMMA GEMM: C[M,N] = A[M,K](bf16) * Bt[N,K](bf16)^T + bias ----------------
// block tile 128x128, 8 waves as 4(M) x 2(N), wave tile 32x64 (2x4 WMMA frags).
// Double-buffered LDS; tiles staged with async global->LDS b128 (ASYNCcnt), with the
// next tile's copies in flight behind the current tile's WMMA work.
__global__ void __launch_bounds__(256) gemm_bf16_wmma(
    const __bf16* __restrict__ A, int lda,
    const __bf16* __restrict__ Bt, int ldb,   // pre-transposed weights [N, K]
    const float* __restrict__ bias,
    float* __restrict__ C, int N, int K) {
  __shared__ __align__(16) __bf16 As[2][128 * 40];   // [m][k], padded
  __shared__ __align__(16) __bf16 Bs[2][128 * 40];   // [n][k], padded
  const int tid = threadIdx.x, lane = tid & 31, wid = tid >> 5;
  const int waveM = wid >> 1, waveN = wid & 1;
  const int m0 = blockIdx.x * 128, n0 = blockIdx.y * 128;

  // per-thread staging coordinates (2 b128 vectors per tile per thread per operand)
  const int sr0 = tid >> 2,          scv0 = (tid & 3) * 8;
  const int sr1 = (tid + 256) >> 2,  scv1 = ((tid + 256) & 3) * 8;

  v8f acc[2][4];
  for (int i = 0; i < 2; ++i)
    for (int j = 0; j < 4; ++j)
      for (int r = 0; r < 8; ++r) acc[i][j][r] = 0.0f;

  const int nk = K >> 5;   // K-steps of 32

  // prologue: stage tile 0 into buffer 0 (4 async instructions per wave)
  async_b128(&As[0][sr0 * 40 + scv0], &A[(size_t)(m0 + sr0) * lda + scv0]);
  async_b128(&Bs[0][sr0 * 40 + scv0], &Bt[(size_t)(n0 + sr0) * ldb + scv0]);
  async_b128(&As[0][sr1 * 40 + scv1], &A[(size_t)(m0 + sr1) * lda + scv1]);
  async_b128(&Bs[0][sr1 * 40 + scv1], &Bt[(size_t)(n0 + sr1) * ldb + scv1]);

  for (int ik = 0; ik < nk; ++ik) {
    const int buf = ik & 1;
    if (ik + 1 < nk) {
      const int k0n = (ik + 1) * 32, nbuf = (ik + 1) & 1;
      async_b128(&As[nbuf][sr0 * 40 + scv0], &A[(size_t)(m0 + sr0) * lda + k0n + scv0]);
      async_b128(&Bs[nbuf][sr0 * 40 + scv0], &Bt[(size_t)(n0 + sr0) * ldb + k0n + scv0]);
      async_b128(&As[nbuf][sr1 * 40 + scv1], &A[(size_t)(m0 + sr1) * lda + k0n + scv1]);
      async_b128(&Bs[nbuf][sr1 * 40 + scv1], &Bt[(size_t)(n0 + sr1) * ldb + k0n + scv1]);
      if (ik + 2 < nk)  // L2 prefetch two tiles ahead
        __builtin_prefetch(&A[(size_t)(m0 + sr0) * lda + (ik + 2) * 32 + scv0], 0, 1);
      wait_async4();   // current tile landed; next tile still in flight
    } else {
      wait_async0();
    }
    __syncthreads();

    const __bf16* as = As[buf];
    const __bf16* bs = Bs[buf];
    v16bf af[2], bfr[4];
    for (int mt = 0; mt < 2; ++mt) af[mt]  = ld_frag(as, waveM * 32 + mt * 16, 0, 40, lane);
    for (int nt = 0; nt < 4; ++nt) bfr[nt] = ld_frag(bs, waveN * 64 + nt * 16, 0, 40, lane);
    for (int mt = 0; mt < 2; ++mt)
      for (int nt = 0; nt < 4; ++nt)
        acc[mt][nt] = __builtin_amdgcn_wmma_f32_16x16x32_bf16(
            false, af[mt], false, bfr[nt], (short)0, acc[mt][nt], false, false);
    __syncthreads();   // all waves' ds reads done before this buffer is re-staged
  }

  // epilogue: + bias, f32 store. C/D layout: VGPR r, lanes<16 -> M=r, lanes>=16 -> M=r+8.
  for (int mt = 0; mt < 2; ++mt)
    for (int nt = 0; nt < 4; ++nt) {
      int mbase = m0 + waveM * 32 + mt * 16 + 8 * (lane >> 4);
      int n     = n0 + waveN * 64 + nt * 16 + (lane & 15);
      float bi = bias[n];
      for (int r = 0; r < 8; ++r)
        C[(size_t)(mbase + r) * N + n] = acc[mt][nt][r] + bi;
    }
}

// ---------------- fused VQ: WMMA scores + per-row argmin + gather + histogram ----------------
__global__ void __launch_bounds__(256) vq_kernel(
    const __bf16* __restrict__ zeb,   // [B,128] bf16
    const __bf16* __restrict__ embb,  // [4096,128] bf16 (already B^T layout)
    const float*  __restrict__ emb,   // [4096,128] f32
    const float*  __restrict__ enorm, // [4096]
    const float*  __restrict__ zef,   // [B,128] f32 (z_e in d_out)
    float* __restrict__ idx_out,      // [B] float-coded indices
    float* __restrict__ zq_out,       // [B,128] f32
    __bf16* __restrict__ zqb,         // [B,128] bf16
    float* __restrict__ counts,       // [4096]
    float* __restrict__ commit_accum) {
  extern __shared__ __align__(16) char smem[];
  __bf16* zeT  = (__bf16*)smem;                   // 128 x 136
  __bf16* embT = zeT + 128 * 136;                 // 128 x 136 (codes x D)
  float*  sc   = (float*)(embT + 128 * 136);      // 128 x 128 scores

  const int tid = threadIdx.x, lane = tid & 31, wid = tid >> 5;
  const int waveM = wid >> 1, waveN = wid & 1;
  const int rowBase = blockIdx.x * 128;

  // stage z_e tile 128x128 via async copies (16 vecs/row, 8 vecs per thread)
  for (int i = 0; i < 8; ++i) {
    int v = tid + 256 * i;
    int r = v >> 4, cv = (v & 15) * 8;
    async_b128(&zeT[r * 136 + cv], &zeb[(size_t)(rowBase + r) * DD + cv]);
  }

  float bestVal = 3.4e38f;
  int   bestIdx = 0;
  const int myRow   = tid & 127;
  const int myCBase = (tid >> 7) * 64;   // threads 0-127 scan codes 0-63, 128-255 scan 64-127

  for (int ct = 0; ct < KCODE / 128; ++ct) {
    const int cgBase = ct * 128;
    __syncthreads();  // previous scan finished; embT free
    for (int i = 0; i < 8; ++i) {
      int v = tid + 256 * i;
      int r = v >> 4, cv = (v & 15) * 8;
      async_b128(&embT[r * 136 + cv], &embb[(size_t)(cgBase + r) * DD + cv]);
    }
    wait_async0();    // covers zeT on first iteration too
    __syncthreads();  // tiles ready

    v8f acc[2][4];
    for (int i = 0; i < 2; ++i)
      for (int j = 0; j < 4; ++j)
        for (int r = 0; r < 8; ++r) acc[i][j][r] = 0.0f;

    for (int kc = 0; kc < DD; kc += 32) {
      v16bf af[2], bfr[4];
      for (int mt = 0; mt < 2; ++mt) af[mt]  = ld_frag(zeT,  waveM * 32 + mt * 16, kc, 136, lane);
      for (int nt = 0; nt < 4; ++nt) bfr[nt] = ld_frag(embT, waveN * 64 + nt * 16, kc, 136, lane);
      for (int mt = 0; mt < 2; ++mt)
        for (int nt = 0; nt < 4; ++nt)
          acc[mt][nt] = __builtin_amdgcn_wmma_f32_16x16x32_bf16(
              false, af[mt], false, bfr[nt], (short)0, acc[mt][nt], false, false);
    }
    // dump raw dot products s(m, n) to LDS
    for (int mt = 0; mt < 2; ++mt)
      for (int nt = 0; nt < 4; ++nt) {
        int mb = waveM * 32 + mt * 16 + 8 * (lane >> 4);
        int n  = waveN * 64 + nt * 16 + (lane & 15);
        for (int r = 0; r < 8; ++r) sc[(mb + r) * 128 + n] = acc[mt][nt][r];
      }
    __syncthreads();  // scores ready

    // running argmin: d2 = ||e||^2 - 2*s  (||z||^2 constant per row)
    for (int j = 0; j < 64; ++j) {
      int cl = myCBase + j;
      float d2v = enorm[cgBase + cl] - 2.0f * sc[myRow * 128 + cl];
      if (d2v < bestVal) { bestVal = d2v; bestIdx = cgBase + cl; }
    }
  }

  __syncthreads();
  float* bv = sc;                 // reuse score buffer for reduction
  int*   bi = (int*)(sc + 256);
  bv[tid] = bestVal;
  bi[tid] = bestIdx;
  __syncthreads();

  if (tid < 128) {
    float v2 = bv[tid + 128];
    int   i2 = bi[tid + 128];
    if (v2 < bestVal || (v2 == bestVal && i2 < bestIdx)) { bestVal = v2; bestIdx = i2; }
    int gr = rowBase + tid;
    idx_out[gr] = (float)bestIdx;
    atomicAdd(&counts[bestIdx], 1.0f);
    const float* er = emb + (size_t)bestIdx * DD;
    float cs = 0.f;
    for (int j = 0; j < DD; ++j) {
      float q = er[j];
      zq_out[(size_t)gr * DD + j] = q;
      zqb[(size_t)gr * DD + j]    = (__bf16)q;
      float dz = q - zef[(size_t)gr * DD + j];
      cs += dz * dz;
    }
    bv[tid] = cs;   // tid<128 writes [0,128), reads above were [128,256): no race
  }
  __syncthreads();
  if (tid == 0) {
    float t = 0.f;
    for (int i = 0; i < 128; ++i) t += bv[i];
    atomicAdd(commit_accum, t);
  }
}

// ---------------- head: [B,128] bf16 @ W6[128,2] + b6 -> d_out velocity ----------------
__global__ void head2_kernel(const __bf16* __restrict__ d2, const float* __restrict__ W6,
                             const float* __restrict__ b6, float* __restrict__ out) {
  int b = blockIdx.x * blockDim.x + threadIdx.x;
  if (b >= BN_) return;
  const __bf16* r = d2 + (size_t)b * DD;
  float a0 = b6[0], a1 = b6[1];
  for (int k = 0; k < DD; ++k) {
    float v = (float)r[k];
    a0 += v * W6[k * 2 + 0];
    a1 += v * W6[k * 2 + 1];
  }
  out[(size_t)b * 2 + 0] = a0;
  out[(size_t)b * 2 + 1] = a1;
}

// ---------------- finalize: commitment_loss, perplexity, n_active ----------------
__global__ void finalize_kernel(const float* __restrict__ counts,
                                const float* __restrict__ scal,
                                float* __restrict__ out) {
  __shared__ float sH[256];
  __shared__ float sA[256];
  float H = 0.f, act = 0.f;
  for (int c = threadIdx.x; c < KCODE; c += 256) {
    float p = counts[c] * (1.0f / (float)BN_);
    H   += -p * logf(p + 1e-10f);
    act += (p > 0.001f) ? 1.0f : 0.0f;
  }
  sH[threadIdx.x] = H; sA[threadIdx.x] = act;
  __syncthreads();
  for (int s = 128; s > 0; s >>= 1) {
    if (threadIdx.x < s) { sH[threadIdx.x] += sH[threadIdx.x + s]; sA[threadIdx.x] += sA[threadIdx.x + s]; }
    __syncthreads();
  }
  if (threadIdx.x == 0) {
    out[OFF_CL] = 0.25f * scal[0] / ((float)BN_ * (float)DD);
    out[OFF_PP] = expf(sH[0]);
    out[OFF_NA] = sA[0];
  }
}

// ---------------- host-side launch ----------------
extern "C" void kernel_launch(void* const* d_in, const int* in_sizes, int n_in,
                              void* d_out, int out_size, void* d_ws, size_t ws_size,
                              hipStream_t stream) {
  (void)in_sizes; (void)n_in; (void)out_size; (void)ws_size;
  const float* x   = (const float*)d_in[0];
  const float* W1  = (const float*)d_in[1];
  const float* b1  = (const float*)d_in[2];
  const float* g1  = (const float*)d_in[3];
  const float* be1 = (const float*)d_in[4];
  const float* W2  = (const float*)d_in[5];
  const float* b2  = (const float*)d_in[6];
  const float* g2  = (const float*)d_in[7];
  const float* be2 = (const float*)d_in[8];
  const float* W3  = (const float*)d_in[9];
  const float* b3  = (const float*)d_in[10];
  const float* emb = (const float*)d_in[11];
  const float* W4  = (const float*)d_in[12];
  const float* b4  = (const float*)d_in[13];
  const float* g3  = (const float*)d_in[14];
  const float* be3 = (const float*)d_in[15];
  const float* W5  = (const float*)d_in[16];
  const float* b5  = (const float*)d_in[17];
  const float* W6  = (const float*)d_in[18];
  const float* b6  = (const float*)d_in[19];
  float* out = (float*)d_out;

  // workspace carve-out
  char* ws = (char*)d_ws;
  size_t off = 0;
  auto alloc = [&](size_t bytes) -> void* {
    void* p = ws + off;
    off = (off + bytes + 255) & ~(size_t)255;
    return p;
  };
  __bf16* XB   = (__bf16*)alloc((size_t)BN_ * K1P * 2);
  __bf16* W1T  = (__bf16*)alloc((size_t)N1 * K1P * 2);   // [512,1440]
  __bf16* W2T  = (__bf16*)alloc((size_t)N2 * N1 * 2);    // [256,512]
  __bf16* W3T  = (__bf16*)alloc((size_t)DD * N2 * 2);    // [128,256]
  __bf16* W4T  = (__bf16*)alloc((size_t)N2 * DD * 2);    // [256,128]
  __bf16* W5T  = (__bf16*)alloc((size_t)DD * N2 * 2);    // [128,256]
  __bf16* EMBB = (__bf16*)alloc((size_t)KCODE * DD * 2);
  float*  ACTF = (float*) alloc((size_t)BN_ * N1 * 4);   // reused f32 activations
  __bf16* ACTB = (__bf16*)alloc((size_t)BN_ * N1 * 2);   // reused bf16 activations
  __bf16* ZEB  = (__bf16*)alloc((size_t)BN_ * DD * 2);
  __bf16* ZQB  = (__bf16*)alloc((size_t)BN_ * DD * 2);
  float*  ENRM = (float*) alloc((size_t)KCODE * 4);
  float*  CNTS = (float*) alloc((size_t)KCODE * 4);
  float*  SCAL = (float*) alloc(64 * 4);

  const int TPB = 256;
  auto gs = [](int n) { int b = (n + 255) / 256; return b > 4096 ? 4096 : b; };

  // ---- prep: casts + weight transposes, zeros, norms ----
  pack_x_kernel<<<4096, TPB, 0, stream>>>(x, XB);
  cast_transpose_kernel<<<gs(N1 * K1P), TPB, 0, stream>>>(W1, W1T, K1, N1, K1P);
  cast_transpose_kernel<<<gs(N2 * N1), TPB, 0, stream>>>(W2, W2T, N1, N2, N1);
  cast_transpose_kernel<<<gs(DD * N2), TPB, 0, stream>>>(W3, W3T, N2, DD, N2);
  cast_transpose_kernel<<<gs(N2 * DD), TPB, 0, stream>>>(W4, W4T, DD, N2, DD);
  cast_transpose_kernel<<<gs(DD * N2), TPB, 0, stream>>>(W5, W5T, N2, DD, N2);
  cast_kernel<<<gs(KCODE * DD), TPB, 0, stream>>>(emb, EMBB, KCODE * DD);
  zero_f32_kernel<<<16, TPB, 0, stream>>>(CNTS, KCODE);
  zero_f32_kernel<<<1, 64, 0, stream>>>(SCAL, 64);
  emb_norms_kernel<<<KCODE / TPB, TPB, 0, stream>>>(emb, ENRM);

  // ---- encoder ----
  gemm_bf16_wmma<<<dim3(BN_ / 128, N1 / 128), TPB, 0, stream>>>(XB, K1P, W1T, K1P, b1, ACTF, N1, K1P);
  ln_gelu_kernel<<<BN_ / 8, TPB, 0, stream>>>(ACTF, g1, be1, ACTB, N1);
  gemm_bf16_wmma<<<dim3(BN_ / 128, N2 / 128), TPB, 0, stream>>>(ACTB, N1, W2T, N1, b2, ACTF, N2, N1);
  ln_gelu_kernel<<<BN_ / 8, TPB, 0, stream>>>(ACTF, g2, be2, ACTB, N2);
  gemm_bf16_wmma<<<dim3(BN_ / 128, DD / 128), TPB, 0, stream>>>(ACTB, N2, W3T, N2, b3, out + OFF_ZE, DD, N2);
  cast_kernel<<<gs(BN_ * DD), TPB, 0, stream>>>(out + OFF_ZE, ZEB, BN_ * DD);

  // ---- VQ (fused scores GEMM + argmin + gather + histogram) ----
  const size_t VQ_SMEM = (size_t)2 * 128 * 136 * 2 + (size_t)128 * 128 * 4;  // 135168 B
  vq_kernel<<<BN_ / 128, TPB, VQ_SMEM, stream>>>(ZEB, EMBB, emb, ENRM, out + OFF_ZE,
                                                 out + OFF_IDX, out + OFF_ZQ, ZQB,
                                                 CNTS, SCAL);

  // ---- decoder ----
  gemm_bf16_wmma<<<dim3(BN_ / 128, N2 / 128), TPB, 0, stream>>>(ZQB, DD, W4T, DD, b4, ACTF, N2, DD);
  ln_gelu_kernel<<<BN_ / 8, TPB, 0, stream>>>(ACTF, g3, be3, ACTB, N2);
  gemm_bf16_wmma<<<dim3(BN_ / 128, DD / 128), TPB, 0, stream>>>(ACTB, N2, W5T, N2, b5, ACTF, DD, N2);
  gelu_cast_kernel<<<gs(BN_ * DD), TPB, 0, stream>>>(ACTF, ACTB, BN_ * DD);
  head2_kernel<<<BN_ / TPB, TPB, 0, stream>>>(ACTB, W6, b6, out + OFF_VEL);

  // ---- scalars ----
  finalize_kernel<<<1, TPB, 0, stream>>>(CNTS, SCAL, out);
}
